// Group_transformer_43568148251443
// MI455X (gfx1250) — matrix operand order repaired and checked by reading, not compiled, not run on hardware
//
#include <hip/hip_runtime.h>

// ---------------------------------------------------------------------------
// Group transformer for MI455X (gfx1250, wave32, WMMA bf16 16x16x32).
// B=8, D=256, M=2048, DT=64, KT=16, UP=4.
// All GEMMs on the bf16 WMMA path (fp32 accumulate). The (B,DT,M,K) attention
// pipeline is fully fused (one wave per query; KT=16 == WMMA N dimension) so
// the 512MB-per-tensor intermediates never touch HBM. LDS buffers are stored
// K-contiguous so every B fragment is 2x ds_load_b128; UP-invariant fragments
// are hoisted into VGPRs.
// ---------------------------------------------------------------------------

typedef __attribute__((ext_vector_type(16))) __bf16 v16bf;
typedef __attribute__((ext_vector_type(8)))  float  v8f;
typedef __attribute__((ext_vector_type(4)))  unsigned int v4u;
typedef unsigned int __attribute__((may_alias)) u32a;
typedef v4u __attribute__((may_alias)) v4ua;

union BF16x16 {
    v16bf v;
    v4u   q[2];
    unsigned int u[8];
    unsigned short s[16];
};

__device__ __forceinline__ unsigned short f2bf(float x) {
    unsigned int u = __float_as_uint(x);
    u += 0x7FFFu + ((u >> 16) & 1u);   // round-to-nearest-even
    return (unsigned short)(u >> 16);
}
__device__ __forceinline__ float bf2f(unsigned short s) {
    return __uint_as_float(((unsigned int)s) << 16);
}
__device__ __forceinline__ unsigned int pack2bf(float a, float b) {
    return (unsigned int)f2bf(a) | ((unsigned int)f2bf(b) << 16);
}

__device__ __forceinline__ v8f wmma_bf16(v16bf a, v16bf b, v8f c) {
    return __builtin_amdgcn_wmma_f32_16x16x32_bf16(false, a, false, b,
                                                   (short)0, c, false, false);
}

// A-matrix (16x32 bf16) fragment from row-major [O x ld] weights (global).
// ISA layout: lane<16 -> K {k0+0..7, k0+16..23}; lane>=16 -> K {k0+8..15, k0+24..31}.
// Two contiguous 16B chunks -> 2x b128 loads.
__device__ __forceinline__ v16bf load_a16(const unsigned short* __restrict__ W,
                                          int ld, int row0, int k0, int lane) {
    const unsigned short* p =
        W + (size_t)(row0 + (lane & 15)) * ld + k0 + ((lane & 16) ? 8 : 0);
    BF16x16 f;
    f.q[0] = *(const v4ua*)p;
    f.q[1] = *(const v4ua*)(p + 16);
    return f.v;
}

// B-matrix (32x16 bf16) fragment from K-contiguous (transposed) LDS buffer:
// X[n * stride + k], stride*2 bytes must be a multiple of 16.
// ISA layout: lane<16 -> K {k0+0..7, k0+16..23}; lane>=16 -> K {k0+8..15, k0+24..31}.
__device__ __forceinline__ v16bf load_bt(const unsigned short* X, int stride,
                                         int k0, int lane) {
    const unsigned short* p =
        X + (lane & 15) * stride + k0 + ((lane & 16) ? 8 : 0);
    BF16x16 f;
    f.q[0] = *(const v4ua*)p;
    f.q[1] = *(const v4ua*)(p + 16);
    return f.v;
}

// ---------------------------------------------------------------------------
// fp32 -> bf16-bits conversion kernels
// ---------------------------------------------------------------------------
__global__ void k_cvt(const float* __restrict__ s, unsigned short* __restrict__ d, int n) {
    int i = blockIdx.x * 256 + threadIdx.x;
    if (i < n) d[i] = f2bf(s[i]);
}

// cat = concat(fts_q, fts_k) along channel, converted to bf16: (B, 512, M)
__global__ void k_cat(const float* __restrict__ q, const float* __restrict__ k,
                      unsigned short* __restrict__ cat) {
    int i = blockIdx.x * 256 + threadIdx.x;               // over 8*512*2048
    int m = i & 2047;
    int c = (i >> 11) & 511;
    int b = i >> 20;
    float v = (c < 256) ? q[((size_t)b * 256 + c) * 2048 + m]
                        : k[((size_t)b * 256 + (c - 256)) * 2048 + m];
    cat[i] = f2bf(v);
}

// ---------------------------------------------------------------------------
// Generic bf16 WMMA GEMM: Y[b] = W (OxC) * X[b] (CxM) + bias, with flags.
// Block = 128 threads (4 waves); 64x64 output tile; transposed LDS staging
// (K-contiguous) so B fragments are 2x ds_load_b128.
// flags: 1 = relu, 2 = accumulate into existing Yf.
// ---------------------------------------------------------------------------
#define GX_STRIDE 40   // u16 elements per column; 80B = 5*16B

__global__ __launch_bounds__(128) void k_gemm(
    const unsigned short* __restrict__ W, const float* __restrict__ bias,
    const unsigned short* __restrict__ X, long xbatch,
    float* __restrict__ Yf, unsigned short* __restrict__ Yh,
    long ybatch, long ybatch_h,
    int C, int Mld, int ldy, int coloff, int flags) {
    __shared__ unsigned short sXt[64 * GX_STRIDE];   // [col][k], 5120B
    int wave = threadIdx.x >> 5, lane = threadIdx.x & 31;
    int b  = blockIdx.z;
    int m0 = blockIdx.x * 64;
    int o0 = blockIdx.y * 64 + wave * 16;
    const unsigned short* Xb = X + (size_t)b * xbatch;

    int r0 = (threadIdx.x & 15) * 2;      // k-row pair for staging
    int c0 = (threadIdx.x >> 4) * 8;      // 8-column group for staging

    v8f acc[4];
#pragma unroll
    for (int t = 0; t < 4; ++t)
#pragma unroll
        for (int j = 0; j < 8; ++j) acc[t][j] = 0.0f;

    for (int kc = 0; kc < C; kc += 32) {
        __syncthreads();
        {   // stage transposed: 2x b128 global loads, 8x packed b32 LDS stores
            v4u va = *(const v4ua*)&Xb[(size_t)(kc + r0) * Mld + m0 + c0];
            v4u vb = *(const v4ua*)&Xb[(size_t)(kc + r0 + 1) * Mld + m0 + c0];
#pragma unroll
            for (int q = 0; q < 4; ++q) {
                unsigned int lo = va[q], hi = vb[q];
                *(u32a*)&sXt[(c0 + 2 * q) * GX_STRIDE + r0] =
                    (lo & 0xFFFFu) | (hi << 16);
                *(u32a*)&sXt[(c0 + 2 * q + 1) * GX_STRIDE + r0] =
                    (lo >> 16) | (hi & 0xFFFF0000u);
            }
        }
        __syncthreads();
        v16bf a = load_a16(W, C, o0, kc, lane);
#pragma unroll
        for (int t = 0; t < 4; ++t) {
            v16bf bb = load_bt(sXt + t * 16 * GX_STRIDE, GX_STRIDE, 0, lane);
            acc[t] = wmma_bf16(a, bb, acc[t]);
        }
    }

    int n = lane & 15, hi = (lane & 16) ? 8 : 0;
#pragma unroll
    for (int t = 0; t < 4; ++t) {
#pragma unroll
        for (int j = 0; j < 8; ++j) {
            int orow = o0 + hi + j;
            int mcol = m0 + t * 16 + n;
            float v = acc[t][j] + bias[orow];
            if (flags & 1) v = fmaxf(v, 0.0f);
            if (Yf) {
                size_t oi = (size_t)b * ybatch + (size_t)orow * ldy + coloff + mcol;
                if (flags & 2) v += Yf[oi];
                Yf[oi] = v;
            }
            if (Yh) Yh[(size_t)b * ybatch_h + (size_t)orow * Mld + mcol] = f2bf(v);
        }
    }
}

// ---------------------------------------------------------------------------
// Brute-force 16-NN: one wave per query, LDS-tiled points, per-lane sorted
// top-16, wave-wide shfl_xor argmin merge (wave32).
// ---------------------------------------------------------------------------
__global__ __launch_bounds__(128) void k_knn(const float* __restrict__ xyz,
                                             int* __restrict__ ids) {
    const int M = 2048;
    __shared__ float sp[3][256];
    int wave = threadIdx.x >> 5, lane = threadIdx.x & 31;
    int b = blockIdx.y;
    int m = blockIdx.x * 4 + wave;
    float qx = xyz[((size_t)b * M + m) * 3 + 0];
    float qy = xyz[((size_t)b * M + m) * 3 + 1];
    float qz = xyz[((size_t)b * M + m) * 3 + 2];

    float bd[16]; int bi[16];
#pragma unroll
    for (int i = 0; i < 16; ++i) { bd[i] = 3.0e38f; bi[i] = -1; }

    for (int t0 = 0; t0 < M; t0 += 256) {
        __syncthreads();
        for (int e = threadIdx.x; e < 256; e += 128) {
            size_t base = ((size_t)b * M + t0 + e) * 3;
            sp[0][e] = xyz[base + 0];
            sp[1][e] = xyz[base + 1];
            sp[2][e] = xyz[base + 2];
        }
        __syncthreads();
        for (int e = lane; e < 256; e += 32) {
            float dx = sp[0][e] - qx, dy = sp[1][e] - qy, dz = sp[2][e] - qz;
            float d = dx * dx + dy * dy + dz * dz;
            if (d < bd[15]) {
                int p = 15;
                while (p > 0 && bd[p - 1] > d) {
                    bd[p] = bd[p - 1]; bi[p] = bi[p - 1]; --p;
                }
                bd[p] = d; bi[p] = e + t0;
            }
        }
    }

    int ptr = 0;
    for (int sel = 0; sel < 16; ++sel) {
        float d  = (ptr < 16) ? bd[ptr] : 3.0e38f;
        int   id = (ptr < 16) ? bi[ptr] : -1;
        float wd = d; int wi = id;
#pragma unroll
        for (int off = 16; off >= 1; off >>= 1) {
            float od = __shfl_xor(wd, off);
            int   oi = __shfl_xor(wi, off);
            if (od < wd || (od == wd && oi < wi)) { wd = od; wi = oi; }
        }
        if (ptr < 16 && id == wi) ++ptr;   // winner lane consumes its head
        if (lane == 0) ids[((size_t)b * M + m) * 16 + sel] = wi;
    }
}

// ---------------------------------------------------------------------------
// Fused attention. Per wave (= one query m):
//   gather k/v via ids, pos MLP (WMMA), build UP-invariant a = q-kl+pos and
//   vp = vl+pos ONCE, then UP x { wa1 GEMM(256x16,K=64) -> relu -> wa2
//   GEMM(64x16,K=256) -> softmax over 16 neighbors -> weighted sum }.
// All LDS matrices K-contiguous; invariant B fragments kept in VGPRs.
// ---------------------------------------------------------------------------
#define H_STRIDE  264   // 528B = 33*16B
#define AB_STRIDE 72    // 144B = 9*16B

struct __align__(16) WaveBuf {
    union {
        unsigned short h[16 * H_STRIDE];            // [n][k], stage1 out / stage2 in
        struct { float kl[1024]; float vl[1024]; } g;  // gather scratch (f32 [dt][n])
    } u;                                            // 8448B
    unsigned short ab[16 * AB_STRIDE];              // [n][dt]: p1, then a  (2304B)
    unsigned short posf[1024];                      // [dt][n] bf16 pos     (2048B)
    unsigned short vp[1024];                        // [dt][n] bf16 vl+pos  (2048B)
    float qv[64];
    float x3[48];
    int   ids[16];
};

__global__ __launch_bounds__(128) void k_attn(
    const float* __restrict__ xyz, const int* __restrict__ ids,
    const float* __restrict__ qf, const float* __restrict__ kf,
    const float* __restrict__ vf,
    const unsigned short* __restrict__ wp2b, const float* __restrict__ bp2,
    const float* __restrict__ wp1, const float* __restrict__ bp1,
    const unsigned short* __restrict__ wa1b, const float* __restrict__ ba1,
    const unsigned short* __restrict__ wa2b, const float* __restrict__ ba2,
    unsigned short* __restrict__ fout) {
    const int M = 2048;
    __shared__ WaveBuf sb[4];
    int wave = threadIdx.x >> 5, lane = threadIdx.x & 31;
    int b = blockIdx.y, m = blockIdx.x * 4 + wave;
    WaveBuf& S = sb[wave];
    int n = lane & 15, hi = (lane & 16) ? 8 : 0;

    if (lane < 16) S.ids[lane] = ids[((size_t)b * M + m) * 16 + lane];
    for (int t = lane; t < 64; t += 32) S.qv[t] = qf[((size_t)b * 64 + t) * M + m];
    __syncthreads();

    if (lane < 16) {
        int id = S.ids[lane];
#pragma unroll
        for (int c = 0; c < 3; ++c)
            S.x3[c * 16 + lane] = xyz[((size_t)b * M + id) * 3 + c] -
                                  xyz[((size_t)b * M + m)  * 3 + c];
    }
    for (int e = lane; e < 1024; e += 32) {      // gather k,v (f32)
        int dt = e >> 4, j = e & 15;
        int id = S.ids[j];
        S.u.g.kl[e] = kf[((size_t)b * 64 + dt) * M + id];
        S.u.g.vl[e] = vf[((size_t)b * 64 + dt) * M + id];
    }
    __syncthreads();

    // p1 = relu(wp1 * pos3 + bp1) -> ab[n][dt] (K=3, VALU), packed stores
    for (int e = lane; e < 512; e += 32) {
        int j  = e >> 5;            // neighbor
        int r0 = (e & 31) * 2;      // dt pair
        float s0 = bp1[r0] + wp1[r0 * 3] * S.x3[j] +
                   wp1[r0 * 3 + 1] * S.x3[16 + j] + wp1[r0 * 3 + 2] * S.x3[32 + j];
        float s1 = bp1[r0 + 1] + wp1[(r0 + 1) * 3] * S.x3[j] +
                   wp1[(r0 + 1) * 3 + 1] * S.x3[16 + j] +
                   wp1[(r0 + 1) * 3 + 2] * S.x3[32 + j];
        *(u32a*)&S.ab[j * AB_STRIDE + r0] =
            pack2bf(fmaxf(s0, 0.0f), fmaxf(s1, 0.0f));
    }

    // pos = wp2 * p1 + bp2 (64x16, K=64) via WMMA; B fragments hoisted
    {
        v16bf pb0 = load_bt(S.ab, AB_STRIDE, 0, lane);
        v16bf pb1 = load_bt(S.ab, AB_STRIDE, 32, lane);
#pragma unroll
        for (int t = 0; t < 4; ++t) {
            v8f acc;
#pragma unroll
            for (int j = 0; j < 8; ++j) acc[j] = 0.0f;
            acc = wmma_bf16(load_a16(wp2b, 64, t * 16, 0, lane), pb0, acc);
            acc = wmma_bf16(load_a16(wp2b, 64, t * 16, 32, lane), pb1, acc);
#pragma unroll
            for (int j = 0; j < 8; ++j) {
                int row = t * 16 + hi + j;
                S.posf[row * 16 + n] = f2bf(acc[j] + bp2[row]);
            }
        }
    }

    // UP-invariant: a = q - kl + pos -> ab[n][dt];  vp = vl + pos -> vp[dt][n]
    for (int e = lane; e < 512; e += 32) {
        int nn = e >> 5;
        int d0 = (e & 31) * 2;
        float v0 = S.qv[d0]     - S.u.g.kl[d0 * 16 + nn]       + bf2f(S.posf[d0 * 16 + nn]);
        float v1 = S.qv[d0 + 1] - S.u.g.kl[(d0 + 1) * 16 + nn] + bf2f(S.posf[(d0 + 1) * 16 + nn]);
        *(u32a*)&S.ab[nn * AB_STRIDE + d0] = pack2bf(v0, v1);
    }
    for (int e = lane; e < 512; e += 32) {
        int dt = e >> 3;
        int n0 = (e & 7) * 2;
        float v0 = S.u.g.vl[dt * 16 + n0]     + bf2f(S.posf[dt * 16 + n0]);
        float v1 = S.u.g.vl[dt * 16 + n0 + 1] + bf2f(S.posf[dt * 16 + n0 + 1]);
        *(u32a*)&S.vp[dt * 16 + n0] = pack2bf(v0, v1);
    }
    __syncthreads();

    // Stage-1 B fragments: invariant across all 4 UP iterations and 16 tiles.
    v16bf ab0 = load_bt(S.ab, AB_STRIDE, 0, lane);
    v16bf ab1 = load_bt(S.ab, AB_STRIDE, 32, lane);

    const float scale = 0.125f;   // 1/sqrt(64)
#pragma unroll 1
    for (int i = 0; i < 4; ++i) {
        const unsigned short* wa1i = wa1b + i * 256 * 64;
        const unsigned short* wa2i = wa2b + i * 64 * 256;
        const float* ba1i = ba1 + i * 256;
        const float* ba2i = ba2 + i * 64;

        // stage 1: h = relu(wa1 * a + ba1), 256x16, K=64 -> h[n][k]
#pragma unroll 1
        for (int t = 0; t < 16; ++t) {
            v8f acc;
#pragma unroll
            for (int j = 0; j < 8; ++j) acc[j] = 0.0f;
            acc = wmma_bf16(load_a16(wa1i, 64, t * 16, 0, lane), ab0, acc);
            acc = wmma_bf16(load_a16(wa1i, 64, t * 16, 32, lane), ab1, acc);
#pragma unroll
            for (int j = 0; j < 8; ++j) {
                int row = t * 16 + hi + j;
                S.u.h[n * H_STRIDE + row] = f2bf(fmaxf(acc[j] + ba1i[row], 0.0f));
            }
        }

        // stage 2: a2 = wa2 * h + ba2 (64x16, K=256); h fragments hoisted
        v16bf hb[8];
#pragma unroll
        for (int s = 0; s < 8; ++s) hb[s] = load_bt(S.u.h, H_STRIDE, s * 32, lane);

#pragma unroll 1
        for (int t = 0; t < 4; ++t) {
            v8f acc;
#pragma unroll
            for (int j = 0; j < 8; ++j) acc[j] = 0.0f;
#pragma unroll
            for (int s = 0; s < 8; ++s)
                acc = wmma_bf16(load_a16(wa2i, 256, t * 16, s * 32, lane), hb[s], acc);
#pragma unroll
            for (int j = 0; j < 8; ++j) {
                int row = t * 16 + hi + j;
                float val = (acc[j] + ba2i[row]) * scale;
                float mx = val;
                mx = fmaxf(mx, __shfl_xor(mx, 1));
                mx = fmaxf(mx, __shfl_xor(mx, 2));
                mx = fmaxf(mx, __shfl_xor(mx, 4));
                mx = fmaxf(mx, __shfl_xor(mx, 8));
                float e = __expf(val - mx);
                float sm = e;
                sm += __shfl_xor(sm, 1);
                sm += __shfl_xor(sm, 2);
                sm += __shfl_xor(sm, 4);
                sm += __shfl_xor(sm, 8);
                float w = (e / sm) * bf2f(S.vp[row * 16 + n]);
                w += __shfl_xor(w, 1);
                w += __shfl_xor(w, 2);
                w += __shfl_xor(w, 4);
                w += __shfl_xor(w, 8);
                if (n == 0)
                    fout[((size_t)(b * 4 + i) * 64 + row) * M + m] = f2bf(w);
            }
        }
        __syncthreads();
    }
}

// ---------------------------------------------------------------------------
// Host side
// ---------------------------------------------------------------------------
extern "C" void kernel_launch(void* const* d_in, const int* in_sizes, int n_in,
                              void* d_out, int out_size, void* d_ws, size_t ws_size,
                              hipStream_t stream) {
    (void)in_sizes; (void)n_in; (void)out_size; (void)ws_size;
    const int B = 8, M = 2048;
    const float* fts_q = (const float*)d_in[0];
    const float* fts_k = (const float*)d_in[1];
    const float* xyz   = (const float*)d_in[2];
    const float* w1    = (const float*)d_in[3];
    const float* b1    = (const float*)d_in[4];
    const float* w2    = (const float*)d_in[5];
    const float* b2    = (const float*)d_in[6];
    const float* wres  = (const float*)d_in[7];
    const float* bres  = (const float*)d_in[8];
    const float* wq    = (const float*)d_in[9];
    const float* bq    = (const float*)d_in[10];
    const float* wk    = (const float*)d_in[11];
    const float* bk    = (const float*)d_in[12];
    const float* wv    = (const float*)d_in[13];
    const float* bv    = (const float*)d_in[14];
    const float* wp1   = (const float*)d_in[15];
    const float* bp1   = (const float*)d_in[16];
    const float* wp2   = (const float*)d_in[17];
    const float* bp2   = (const float*)d_in[18];
    const float* wa1   = (const float*)d_in[19];
    const float* ba1   = (const float*)d_in[20];
    const float* wa2   = (const float*)d_in[21];
    const float* ba2   = (const float*)d_in[22];
    const float* wo    = (const float*)d_in[23];
    const float* bo    = (const float*)d_in[24];
    const float* wr    = (const float*)d_in[25];
    const float* br    = (const float*)d_in[26];
    float* out = (float*)d_out;

    char* ws = (char*)d_ws;
    size_t off = 0;
    auto alloc = [&](size_t bytes) -> void* {
        void* p = ws + off;
        off = (off + bytes + 255) & ~(size_t)255;
        return p;
    };
    unsigned short* catb  = (unsigned short*)alloc((size_t)B * 512 * M * 2);
    unsigned short* Hb    = (unsigned short*)alloc((size_t)B * 256 * M * 2);
    float*          ftsv  = (float*)        alloc((size_t)B * 256 * M * 4);
    unsigned short* ftsvb = (unsigned short*)alloc((size_t)B * 256 * M * 2);
    float*          qbuf  = (float*)        alloc((size_t)B * 64 * M * 4);
    float*          kbuf  = (float*)        alloc((size_t)B * 64 * M * 4);
    float*          vbuf  = (float*)        alloc((size_t)B * 64 * M * 4);
    int*            idsb  = (int*)          alloc((size_t)B * M * 16 * 4);
    unsigned short* fb    = (unsigned short*)alloc((size_t)B * 4 * 64 * M * 2);
    unsigned short* w1b   = (unsigned short*)alloc(256 * 512 * 2);
    unsigned short* w2b   = (unsigned short*)alloc(256 * 256 * 2);
    unsigned short* wresb = (unsigned short*)alloc(256 * 512 * 2);
    unsigned short* wqb   = (unsigned short*)alloc(64 * 256 * 2);
    unsigned short* wkb   = (unsigned short*)alloc(64 * 256 * 2);
    unsigned short* wvb   = (unsigned short*)alloc(64 * 256 * 2);
    unsigned short* wp2b  = (unsigned short*)alloc(64 * 64 * 2);
    unsigned short* wa1b  = (unsigned short*)alloc(4 * 256 * 64 * 2);
    unsigned short* wa2b  = (unsigned short*)alloc(4 * 64 * 256 * 2);
    unsigned short* wob   = (unsigned short*)alloc(4 * 256 * 64 * 2);
    unsigned short* wrb   = (unsigned short*)alloc(4 * 256 * 256 * 2);

    auto cvt = [&](const float* s, unsigned short* d, int nelem) {
        k_cvt<<<(nelem + 255) / 256, 256, 0, stream>>>(s, d, nelem);
    };
    cvt(w1, w1b, 256 * 512);     cvt(w2, w2b, 256 * 256);
    cvt(wres, wresb, 256 * 512); cvt(wq, wqb, 64 * 256);
    cvt(wk, wkb, 64 * 256);      cvt(wv, wvb, 64 * 256);
    cvt(wp2, wp2b, 64 * 64);     cvt(wa1, wa1b, 4 * 256 * 64);
    cvt(wa2, wa2b, 4 * 64 * 256);cvt(wo, wob, 4 * 256 * 64);
    cvt(wr, wrb, 4 * 256 * 256);
    k_cat<<<(B * 512 * M) / 256, 256, 0, stream>>>(fts_q, fts_k, catb);
    k_knn<<<dim3(M / 4, B), 128, 0, stream>>>(xyz, idsb);

    // H = relu(w1*cat + b1)                         (C=512, bf16 out)
    k_gemm<<<dim3(32, 4, B), 128, 0, stream>>>(w1b, b1, catb, (long)512 * M,
        nullptr, Hb, 0, (long)256 * M, 512, M, M, 0, 1);
    // ftsv = wres*cat + bres                        (C=512, f32 out)
    k_gemm<<<dim3(32, 4, B), 128, 0, stream>>>(wresb, bres, catb, (long)512 * M,
        ftsv, nullptr, (long)256 * M, 0, 512, M, M, 0, 0);
    // ftsv += w2*H + b2 ; ftsvb = bf16(ftsv)        (C=256, accumulate)
    k_gemm<<<dim3(32, 4, B), 128, 0, stream>>>(w2b, b2, Hb, (long)256 * M,
        ftsv, ftsvb, (long)256 * M, (long)256 * M, 256, M, M, 0, 2);
    // q/k/v projections (O=64)
    k_gemm<<<dim3(32, 1, B), 128, 0, stream>>>(wqb, bq, catb, (long)512 * M,
        qbuf, nullptr, (long)64 * M, 0, 256, M, M, 0, 0);
    k_gemm<<<dim3(32, 1, B), 128, 0, stream>>>(wkb, bk, catb + 256 * M, (long)512 * M,
        kbuf, nullptr, (long)64 * M, 0, 256, M, M, 0, 0);
    k_gemm<<<dim3(32, 1, B), 128, 0, stream>>>(wvb, bv, ftsvb, (long)256 * M,
        vbuf, nullptr, (long)64 * M, 0, 256, M, M, 0, 0);

    // fused gather + pos MLP + 4x attention MLP + softmax + reduce
    k_attn<<<dim3(M / 4, B), 128, 0, stream>>>(xyz, idsb, qbuf, kbuf, vbuf,
        wp2b, bp2, wp1, bp1, wa1b, ba1, wa2b, ba2, fb);

    // out[:, :, i*M:(i+1)*M] = wr_i*ftsv + br_i + wo_i*f_i + bo_i
    for (int i = 0; i < 4; ++i) {
        k_gemm<<<dim3(32, 4, B), 128, 0, stream>>>(wrb + i * 256 * 256, br + i * 256,
            ftsvb, (long)256 * M, out, nullptr, (long)256 * 4 * M, 0,
            256, M, 4 * M, i * M, 0);
        k_gemm<<<dim3(32, 4, B), 128, 0, stream>>>(wob + i * 256 * 64, bo + i * 256,
            fb + (size_t)i * 64 * M, (long)4 * 64 * M, out, nullptr,
            (long)256 * 4 * M, 0, 64, M, 4 * M, i * M, 2);
    }
}
